// AttentionHead_65206193487967
// MI455X (gfx1250) — matrix operand order repaired and checked by reading, not compiled, not run on hardware
//
#include <hip/hip_runtime.h>
#include <hip/hip_bf16.h>

#define EMBED 768
#define HDIM  64
#define BATCH 8
#define SEQ   2048
#define NTOK  (BATCH * SEQ)           // 16384
#define LOG2E 1.44269504088896340736f

typedef __attribute__((ext_vector_type(16))) _Float16     v16h;
typedef __attribute__((ext_vector_type(8)))  _Float16     v8h;
typedef __attribute__((ext_vector_type(8)))  float        v8f;
typedef __attribute__((ext_vector_type(4)))  unsigned int v4u;
typedef __attribute__((ext_vector_type(8)))  int          v8i;
typedef __attribute__((ext_vector_type(4)))  int          v4i;

// ---- fragment helpers (layouts per CDNA5 ISA 7.12.2, wave32) ------------

__device__ __forceinline__ v16h make16(v8h lo, v8h hi) {
    v16h r;
#pragma unroll
    for (int i = 0; i < 8; ++i) { r[i] = lo[i]; r[i + 8] = hi[i]; }
    return r;
}

// A-matrix 16x32 f16: lane (0-15) row M=lane, K = {0..7, 16..23};
// lanes 16-31 same rows, K = {8..15, 24..31}.
__device__ __forceinline__ v16h loadA(const _Float16* base, int stride, int lane) {
    const _Float16* p = base + (lane & 15) * stride + ((lane & 16) ? 8 : 0);
    return make16(*(const v8h*)p, *(const v8h*)(p + 16));
}

// B-matrix 32x16 f16 from an N-major [16][K] buffer: lane column n = lane&15,
// K = {0..7, 16..23} (lanes 0-15) or {8..15, 24..31} (lanes 16-31).
__device__ __forceinline__ v16h loadB(const _Float16* base, int stride, int lane) {
    const _Float16* p = base + (lane & 15) * stride + ((lane & 16) ? 8 : 0);
    return make16(*(const v8h*)p, *(const v8h*)(p + 16));
}

__device__ __forceinline__ float rmax16(float v) {
    v = fmaxf(v, __shfl_xor(v, 1, 32));
    v = fmaxf(v, __shfl_xor(v, 2, 32));
    v = fmaxf(v, __shfl_xor(v, 4, 32));
    v = fmaxf(v, __shfl_xor(v, 8, 32));
    return v;
}
__device__ __forceinline__ float rsum16(float v) {
    v += __shfl_xor(v, 1, 32);
    v += __shfl_xor(v, 2, 32);
    v += __shfl_xor(v, 4, 32);
    v += __shfl_xor(v, 8, 32);
    return v;
}

#define WMMA_F16(A, B, C) \
    __builtin_amdgcn_wmma_f32_16x16x32_f16(false, (A), false, (B), (short)0, (C), false, false)

// ---- Tensor Data Mover: 2D f16 tile, global -> LDS, with LDS row padding.
// D# layout per CDNA5 ISA (group0 128b + group1 256b).
// pad_interval code: pads after 2^(code+1) dwords; pad_amount code: (code+1) dwords.
// This toolchain exposes the 6-arg builtin: (v4u, v8i, v4i, v4i, v8i, i32 cpol).
__device__ __forceinline__ void tdm_load_2d_f16(
        unsigned int lds_off, const void* gaddr,
        unsigned int tensor_d0, unsigned int tensor_d1, unsigned long long stride0_elems,
        unsigned int tile_d0, unsigned int tile_d1,
        unsigned int pad_interval_code, unsigned int pad_amount_code) {
    unsigned long long ga = (unsigned long long)gaddr;
    v4u g0;
    g0[0] = 1u;                                               // count=1, user D#
    g0[1] = lds_off;                                          // lds_addr (bytes)
    g0[2] = (unsigned int)(ga & 0xFFFFFFFFu);                 // global_addr lo
    g0[3] = (unsigned int)((ga >> 32) & 0x01FFFFFFu) | (2u << 30);  // hi25 | type=2
    v8i g1;
    unsigned int d0 = (1u << 16)                              // data_size = 2 bytes
                    | (1u << 20)                              // pad_enable
                    | (pad_interval_code << 22)
                    | (pad_amount_code << 25);
    g1[0] = (int)d0;
    g1[1] = (int)((tensor_d0 & 0xFFFFu) << 16);               // abar_addr=0 | dim0 lo16
    g1[2] = (int)(((tensor_d0 >> 16) & 0xFFFFu) | ((tensor_d1 & 0xFFFFu) << 16));
    g1[3] = (int)(((tensor_d1 >> 16) & 0xFFFFu) | ((tile_d0 & 0xFFFFu) << 16));
    g1[4] = (int)(tile_d1 & 0xFFFFu);                         // tile_dim1 | tile_dim2=0
    g1[5] = (int)(unsigned int)(stride0_elems & 0xFFFFFFFFu); // dim0_stride lo32
    g1[6] = (int)(unsigned int)((stride0_elems >> 32) & 0xFFFFu); // dim0_stride hi16
    g1[7] = 0;                                                // dim1_stride hi (2D: unused)
    v4i z4 = {0, 0, 0, 0};
    v8i z8 = {0, 0, 0, 0, 0, 0, 0, 0};
    __builtin_amdgcn_tensor_load_to_lds(g0, g1, z4, z4, z8, 0);
}

__device__ __forceinline__ unsigned int lds_offset_of(const void* p) {
    return (unsigned int)(unsigned long long)p;   // generic LDS ptr: offset in [31:0]
}

// =========================================================================
// Kernel 1: fused QKV projection, fp32 -> f16 WMMA -> f16 workspace.
// Q,K stored row-major [token][h]; V stored TRANSPOSED [b][h][s] so the
// attention kernel's V tiles are plain 2D strided tiles for the TDM.
// grid (NTOK/16, 3), block 128 (4 waves; wave w owns output cols 16w..16w+15)
// =========================================================================
__global__ void __launch_bounds__(128)
attn_proj_kernel(const float* __restrict__ xq, const float* __restrict__ xk,
                 const float* __restrict__ xv,
                 const float* __restrict__ Wq, const float* __restrict__ Wk,
                 const float* __restrict__ Wv,
                 const float* __restrict__ bq, const float* __restrict__ bk,
                 const float* __restrict__ bv,
                 _Float16* __restrict__ Qh, _Float16* __restrict__ Kh,
                 _Float16* __restrict__ VtG) {
    __shared__ _Float16 Ah[16 * 40];   // 16 tokens x 32 E-chunk (pad 40)
    __shared__ _Float16 Wt[64 * 40];   // transposed weights: [h][e_local] (pad 40)

    const int tid  = threadIdx.x;
    const int w    = tid >> 5;
    const int lane = tid & 31;
    const int t0   = blockIdx.x * 16;

    const float* X; const float* W; const float* bias;
    if (blockIdx.y == 0)      { X = xq; W = Wq; bias = bq; }
    else if (blockIdx.y == 1) { X = xk; W = Wk; bias = bk; }
    else                      { X = xv; W = Wv; bias = bv; }

    v8f acc = {};

    for (int e0 = 0; e0 < EMBED; e0 += 32) {
        // --- stage input tile 16x32 fp32 -> f16 (512 floats, 4 per thread)
        {
            int idx = tid * 4;
            int row = idx >> 5, col = idx & 31;
            const float* src = X + (size_t)(t0 + row) * EMBED + e0 + col;
            float4 f = *(const float4*)src;
            __builtin_prefetch(src + 32, 0, 1);               // next E-chunk
            Ah[row * 40 + col + 0] = (_Float16)f.x;
            Ah[row * 40 + col + 1] = (_Float16)f.y;
            Ah[row * 40 + col + 2] = (_Float16)f.z;
            Ah[row * 40 + col + 3] = (_Float16)f.w;
        }
        // --- stage weight chunk 32x64 fp32 -> transposed f16 [h][e] (16/thread)
        {
            int base = tid * 16;
            int e = base >> 6, h0 = base & 63;
#pragma unroll
            for (int k = 0; k < 4; ++k) {
                float4 f = *(const float4*)(W + (size_t)(e0 + e) * HDIM + h0 + k * 4);
                Wt[(h0 + k * 4 + 0) * 40 + e] = (_Float16)f.x;
                Wt[(h0 + k * 4 + 1) * 40 + e] = (_Float16)f.y;
                Wt[(h0 + k * 4 + 2) * 40 + e] = (_Float16)f.z;
                Wt[(h0 + k * 4 + 3) * 40 + e] = (_Float16)f.w;
            }
        }
        __syncthreads();

        v16h a = loadA(Ah, 40, lane);
        v16h b = loadB(Wt + w * 16 * 40, 40, lane);
        acc = WMMA_F16(a, b, acc);

        __syncthreads();
    }

    // C layout: VGPR r holds row M = r + 8*(lane>=16), col n = lane&15
    const int n   = lane & 15;
    const int mof = (lane & 16) ? 8 : 0;
    const float bv_ = bias[w * 16 + n];

    if (blockIdx.y == 2) {
        // V: store transposed [b][h][s]
        const int bb = t0 / SEQ;
        const int s0 = t0 % SEQ;
        _Float16* vb = VtG + ((size_t)bb * HDIM + w * 16 + n) * SEQ + s0;
#pragma unroll
        for (int r = 0; r < 8; ++r)
            vb[mof + r] = (_Float16)(acc[r] + bv_);
    } else {
        _Float16* O = (blockIdx.y == 0) ? Qh : Kh;
#pragma unroll
        for (int r = 0; r < 8; ++r)
            O[(size_t)(t0 + mof + r) * HDIM + w * 16 + n] = (_Float16)(acc[r] + bv_);
    }
}

// =========================================================================
// Kernel 2: flash attention with double-buffered TDM tile staging.
// grid (SEQ/64, BATCH), block 128. Wave w owns query rows [64*bx+16w, +16).
// =========================================================================
__global__ void __launch_bounds__(128)
attn_flash_kernel(const _Float16* __restrict__ Qh, const _Float16* __restrict__ Kh,
                  const _Float16* __restrict__ VtG,
                  const float* __restrict__ qmask, const float* __restrict__ kmask,
                  float* __restrict__ out) {
    __shared__ _Float16 Kt[2][32 * 72];    // [token][h], TDM-padded stride 72
    __shared__ _Float16 Vt[2][64 * 40];    // [h][token], TDM-padded stride 40
    __shared__ _Float16 Ps[4 * 16 * 40];   // per-wave P staging, pad 40

    const int tid  = threadIdx.x;
    const int w    = tid >> 5;
    const int lane = tid & 31;
    const int b    = blockIdx.y;
    const int qt0  = blockIdx.x * 64;

    const _Float16* Qb  = Qh  + (size_t)b * SEQ * HDIM;
    const _Float16* Kb  = Kh  + (size_t)b * SEQ * HDIM;
    const _Float16* Vtb = VtG + (size_t)b * HDIM * SEQ;   // [h][s]

    const int n   = lane & 15;
    const int mof = (lane & 16) ? 8 : 0;
    const float alpha = (1.0f / 8.0f) * LOG2E;            // 1/sqrt(64), log2 domain

    // Q fragments resident in registers for the whole pass (h 0..31, 32..63)
    const _Float16* qbase = Qb + (size_t)(qt0 + w * 16) * HDIM;
    v16h a0 = loadA(qbase, HDIM, lane);
    v16h a1 = loadA(qbase + 32, HDIM, lane);

    // per-row query-mask bias (additive -inf, like masked_fill)
    float qmn[8];
#pragma unroll
    for (int r = 0; r < 8; ++r) {
        float qm = qmask[(size_t)b * SEQ + qt0 + w * 16 + mof + r];
        qmn[r] = (qm == 0.0f) ? -__builtin_inff() : 0.0f;
    }

    v8f o0 = {}, o1 = {}, o2 = {}, o3 = {};
    float m[8], l[8];
#pragma unroll
    for (int r = 0; r < 8; ++r) { m[r] = -__builtin_inff(); l[r] = 0.0f; }

    const int NT = SEQ / 32;   // 64 key tiles

    // prologue: DMA tile 0 into buffer 0 (wave 0 drives the TDM)
    if (w == 0) {
        // K tile: 32 rows x 64 halves, row stride HDIM; LDS pad 4dw/32dw -> 72
        tdm_load_2d_f16(lds_offset_of(&Kt[0][0]), Kb, 64, 32, HDIM, 64, 32, 4, 3);
        // V tile: 64 rows x 32 halves, row stride SEQ; LDS pad 4dw/16dw -> 40
        tdm_load_2d_f16(lds_offset_of(&Vt[0][0]), Vtb, 32, 64, SEQ, 32, 64, 3, 3);
    }

    for (int it = 0; it < NT; ++it) {
        const int cur = it & 1;
        const int kt0 = it * 32;

        if (w == 0) {
            if (it + 1 < NT) {
                const int nxt = (it + 1) & 1;
                const int kn  = (it + 1) * 32;
                tdm_load_2d_f16(lds_offset_of(&Kt[nxt][0]), Kb + (size_t)kn * HDIM,
                                64, 32, HDIM, 64, 32, 4, 3);
                tdm_load_2d_f16(lds_offset_of(&Vt[nxt][0]), Vtb + kn,
                                32, 64, SEQ, 32, 64, 3, 3);
                __builtin_amdgcn_s_wait_tensorcnt(2);   // current buffer resident
            } else {
                __builtin_amdgcn_s_wait_tensorcnt(0);
            }
        }
        __syncthreads();

        const _Float16* KtL = &Kt[cur][0];
        const _Float16* VtL = &Vt[cur][0];

        // ---- S = Q K^T for 16q x 32k (4 WMMAs)
        v8f s0 = {}, s1 = {};
        s0 = WMMA_F16(a0, loadB(KtL, 72, lane), s0);
        s0 = WMMA_F16(a1, loadB(KtL + 32, 72, lane), s0);
        s1 = WMMA_F16(a0, loadB(KtL + 16 * 72, 72, lane), s1);
        s1 = WMMA_F16(a1, loadB(KtL + 16 * 72 + 32, 72, lane), s1);

        // ---- scale + masks (additive -inf)
        float km0 = kmask[(size_t)b * SEQ + kt0 + n];
        float km1 = kmask[(size_t)b * SEQ + kt0 + 16 + n];
        float kmn0 = (km0 == 0.0f) ? -__builtin_inff() : 0.0f;
        float kmn1 = (km1 == 0.0f) ? -__builtin_inff() : 0.0f;
#pragma unroll
        for (int r = 0; r < 8; ++r) {
            s0[r] = s0[r] * alpha + qmn[r] + kmn0;
            s1[r] = s1[r] * alpha + qmn[r] + kmn1;
        }

        // ---- online softmax (row = VGPR index; 16-lane group reductions)
        float p0[8], p1[8];
        _Float16* Pl = Ps + w * 16 * 40;
#pragma unroll
        for (int r = 0; r < 8; ++r) {
            float mr = rmax16(fmaxf(s0[r], s1[r]));
            float mn = fmaxf(m[r], mr);
            float corr = __builtin_amdgcn_exp2f(m[r] - mn);
            p0[r] = __builtin_amdgcn_exp2f(s0[r] - mn);
            p1[r] = __builtin_amdgcn_exp2f(s1[r] - mn);
            float rs = rsum16(p0[r] + p1[r]);
            l[r] = l[r] * corr + rs;
            m[r] = mn;
            o0[r] *= corr; o1[r] *= corr; o2[r] *= corr; o3[r] *= corr;
            // stage P (C layout -> row-major [16][32]) for the A-fragment reload
            Pl[(mof + r) * 40 + n]      = (_Float16)p0[r];
            Pl[(mof + r) * 40 + n + 16] = (_Float16)p1[r];
        }
        // wave-local LDS transpose handoff: DS ops are in-order per wave,
        // but fence the compiler and the DS counter explicitly.
        __builtin_amdgcn_wave_barrier();
        asm volatile("s_wait_dscnt 0" ::: "memory");

        v16h pa = loadA(Pl, 40, lane);

        // ---- O += P * V (4 WMMAs over the 4 h-groups)
        o0 = WMMA_F16(pa, loadB(VtL + 0 * 16 * 40, 40, lane), o0);
        o1 = WMMA_F16(pa, loadB(VtL + 1 * 16 * 40, 40, lane), o1);
        o2 = WMMA_F16(pa, loadB(VtL + 2 * 16 * 40, 40, lane), o2);
        o3 = WMMA_F16(pa, loadB(VtL + 3 * 16 * 40, 40, lane), o3);

        __syncthreads();   // all reads done before next TDM overwrites buffers
    }

    // ---- normalize and store fp32 output
    float* ob = out + (size_t)b * SEQ * HDIM;
#pragma unroll
    for (int r = 0; r < 8; ++r) {
        float inv = 1.0f / l[r];
        size_t row = (size_t)(qt0 + w * 16 + mof + r) * HDIM;
        ob[row + 0 * 16 + n] = o0[r] * inv;
        ob[row + 1 * 16 + n] = o1[r] * inv;
        ob[row + 2 * 16 + n] = o2[r] * inv;
        ob[row + 3 * 16 + n] = o3[r] * inv;
    }
}

// =========================================================================
extern "C" void kernel_launch(void* const* d_in, const int* in_sizes, int n_in,
                              void* d_out, int out_size, void* d_ws, size_t ws_size,
                              hipStream_t stream) {
    (void)in_sizes; (void)n_in; (void)out_size; (void)ws_size;
    const float* query = (const float*)d_in[0];
    const float* key   = (const float*)d_in[1];
    const float* value = (const float*)d_in[2];
    const float* qmask = (const float*)d_in[3];
    const float* kmask = (const float*)d_in[4];
    const float* Wq    = (const float*)d_in[5];
    const float* bq    = (const float*)d_in[6];
    const float* Wk    = (const float*)d_in[7];
    const float* bk    = (const float*)d_in[8];
    const float* Wv    = (const float*)d_in[9];
    const float* bv    = (const float*)d_in[10];

    _Float16* Qh  = (_Float16*)d_ws;                // 3 x 2 MB f16 staging in ws
    _Float16* Kh  = Qh + (size_t)NTOK * HDIM;
    _Float16* VtG = Kh + (size_t)NTOK * HDIM;       // V transposed [b][h][s]

    attn_proj_kernel<<<dim3(NTOK / 16, 3), 128, 0, stream>>>(
        query, key, value, Wq, Wk, Wv, bq, bk, bv, Qh, Kh, VtG);

    attn_flash_kernel<<<dim3(SEQ / 64, BATCH), 128, 0, stream>>>(
        Qh, Kh, VtG, qmask, kmask, (float*)d_out);
}
